// PoseModel_31044023616318
// MI455X (gfx1250) — compile-verified, matrix operand order checked by reading, not run on hardware
//
#include <hip/hip_runtime.h>
#include <hip/hip_bf16.h>

typedef _Float16 h8   __attribute__((ext_vector_type(8)));
typedef _Float16 v16h __attribute__((ext_vector_type(16)));
typedef float    v8f  __attribute__((ext_vector_type(8)));

#define BATCH 65536
#define L 33
#define NF 30
#define LA 66          // L * A
#define K1 1089        // L*L
#define KPAD 1120      // 35 * 32
#define KB1 35         // K-blocks of 32 for GEMM1
#define H1 64
#define H2 32
#define ROWS 16        // batch rows per block

// ---------------------------------------------------------------------------
// Kernel 1: convert W1 (1089x64 f32, row-major K x N) and W2 (64x32 f32) to
// f16 in the exact CDNA5 16-bit B-fragment order:
//   per (ntile, kblock, lane): 16 halfs
//     n   = ntile*16 + (lane & 15)
//     khi = (lane >> 4) * 8
//     half[e]   = W[kblock*32 + khi +      e][n]   e = 0..7   (VGPRs 0-3)
//     half[8+e] = W[kblock*32 + khi + 16 + e][n]   e = 0..7   (VGPRs 4-7)
// K is zero-padded 1089 -> 1120 for W1.
// ---------------------------------------------------------------------------
__global__ void pack_weights(const float* __restrict__ W1,
                             const float* __restrict__ W2,
                             _Float16* __restrict__ W1p,
                             _Float16* __restrict__ W2p) {
  int t = blockIdx.x * blockDim.x + threadIdx.x;
  const int T1 = (H1 / 16) * KB1 * 32;     // 4 * 35 * 32 = 4480 fragment-lanes
  const int T2 = (H2 / 16) * 2 * 32;       // 2 *  2 * 32 =  128 fragment-lanes
  if (t < T1) {
    int lane = t & 31;
    int kb   = (t >> 5) % KB1;
    int nt   = (t >> 5) / KB1;
    int n    = nt * 16 + (lane & 15);
    int khi  = (lane >> 4) * 8;
    _Float16* dst = W1p + (size_t)t * 16;
#pragma unroll
    for (int e = 0; e < 8; ++e) {
      int k0 = kb * 32 + khi + e;
      int k1 = k0 + 16;
      dst[e]     = (k0 < K1) ? (_Float16)W1[(size_t)k0 * H1 + n] : (_Float16)0.f;
      dst[8 + e] = (k1 < K1) ? (_Float16)W1[(size_t)k1 * H1 + n] : (_Float16)0.f;
    }
  } else if (t < T1 + T2) {
    int u    = t - T1;
    int lane = u & 31;
    int kb   = (u >> 5) & 1;
    int nt   = (u >> 5) >> 1;
    int n    = nt * 16 + (lane & 15);
    int khi  = (lane >> 4) * 8;
    _Float16* dst = W2p + (size_t)u * 16;
#pragma unroll
    for (int e = 0; e < 8; ++e) {
      int k0 = kb * 32 + khi + e;        // < 64 always
      int k1 = k0 + 16;
      dst[e]     = (_Float16)W2[(size_t)k0 * H2 + n];
      dst[8 + e] = (_Float16)W2[(size_t)k1 * H2 + n];
    }
  }
}

// ---------------------------------------------------------------------------
// Kernel 2: fully fused pose forward for a 16-row batch tile.
// 128 threads = 4 wave32's.
// ---------------------------------------------------------------------------
__global__ void __launch_bounds__(128)
pose_fused(const float* __restrict__ x,
           const float* __restrict__ b1,
           const float* __restrict__ b2,
           const _Float16* __restrict__ W1p,
           const _Float16* __restrict__ W2p,
           float* __restrict__ out) {
  __shared__ float    s_lm[ROWS][LA];      //  4.2 KB landmark means
  __shared__ float    s_vx[ROWS][L];       //  2.1 KB
  __shared__ float    s_vy[ROWS][L];       //  2.1 KB
  __shared__ float    s_n [ROWS][L];       //  2.1 KB
  __shared__ _Float16 s_ang[ROWS][KPAD];   // 35.0 KB angle tile (f16, padded)
  __shared__ _Float16 s_h [ROWS][H1];      //  2.0 KB hidden activations (f16)

  const int tid  = threadIdx.x;
  const int lane = tid & 31;
  const int wave = tid >> 5;
  const long long brow = (long long)blockIdx.x * ROWS;

  // ---- Stage A: frame reduction: lm = sum(x)/count_nonzero(x) over F ----
  for (int t = tid; t < ROWS * LA; t += 128) {
    int r = t / LA, idx = t - r * LA;
    const float* px = x + (brow + r) * (long long)(NF * LA) + idx;
    float s = 0.f, cnt = 0.f;
#pragma unroll
    for (int f = 0; f < NF; ++f) {
      float v = px[f * LA];
      s += v;
      cnt += (v != 0.f) ? 1.f : 0.f;
    }
    s_lm[r][idx] = s / cnt;
  }
  __syncthreads();

  // ---- Stage A2: vectors from landmark 0 + norms ----
  for (int t = tid; t < ROWS * L; t += 128) {
    int r = t / L, l = t - r * L;
    float vx = s_lm[r][l * 2 + 0] - s_lm[r][0];
    float vy = s_lm[r][l * 2 + 1] - s_lm[r][1];
    s_vx[r][l] = vx;
    s_vy[r][l] = vy;
    s_n[r][l]  = sqrtf(vx * vx + vy * vy);
  }
  __syncthreads();

  // ---- Stage B: 33x33 angle matrix -> f16 LDS tile, K zero-padded ----
  for (int t = tid; t < ROWS * KPAD; t += 128) {
    int r = t / KPAD, k = t - r * KPAD;
    float a = 0.f;
    if (k < K1) {
      int i = k / L, j = k - i * L;
      float dot = s_vx[r][i] * s_vx[r][j] + s_vy[r][i] * s_vy[r][j];
      float den = s_n[r][i] * s_n[r][j];
      float c   = dot / ((den > 0.f) ? den : 1.f);
      bool valid = (den > 0.f) && (fabsf(c) <= 1.f);
      a = valid ? acosf(c) : 0.f;
    }
    s_ang[r][k] = (_Float16)a;
  }
  __syncthreads();

  // ---- Stage C: GEMM1 16x1120 @ 1120x64, each wave owns one 16-col N-tile --
  const int m   = lane & 15;            // A row / D column index
  const int khi = (lane >> 4) * 8;      // K-half select per 16-bit A layout
  const int mrb = (lane >> 4) * 8;      // D row base per C/D layout

  {
    const int nt = wave;                // 0..3
    v8f acc = {};
    for (int kb = 0; kb < KB1; ++kb) {
      h8 a0 = *(const h8*)&s_ang[m][kb * 32 + khi];
      h8 a1 = *(const h8*)&s_ang[m][kb * 32 + khi + 16];
      v16h A = __builtin_shufflevector(a0, a1,
                                       0, 1, 2, 3, 4, 5, 6, 7,
                                       8, 9, 10, 11, 12, 13, 14, 15);
      v16h Bf = *(const v16h*)(W1p + ((size_t)(nt * KB1 + kb) * 32 + lane) * 16);
      acc = __builtin_amdgcn_wmma_f32_16x16x32_f16(
          false, A, false, Bf, (short)0, acc, false, false);
    }
    float bias = b1[nt * 16 + (lane & 15)];
#pragma unroll
    for (int r = 0; r < 8; ++r) {
      float h = fmaxf(acc[r] + bias, 0.f);                 // relu(h@W1+b1)
      s_h[mrb + r][nt * 16 + (lane & 15)] = (_Float16)h;   // repack for GEMM2
    }
  }
  __syncthreads();

  // ---- Stage D: GEMM2 16x64 @ 64x32, waves 0/1 own the two 16-col N-tiles --
  if (wave < 2) {
    const int nt = wave;                // 0..1
    v8f acc = {};
#pragma unroll
    for (int kb = 0; kb < 2; ++kb) {
      h8 a0 = *(const h8*)&s_h[m][kb * 32 + khi];
      h8 a1 = *(const h8*)&s_h[m][kb * 32 + khi + 16];
      v16h A = __builtin_shufflevector(a0, a1,
                                       0, 1, 2, 3, 4, 5, 6, 7,
                                       8, 9, 10, 11, 12, 13, 14, 15);
      v16h Bf = *(const v16h*)(W2p + ((size_t)(nt * 2 + kb) * 32 + lane) * 16);
      acc = __builtin_amdgcn_wmma_f32_16x16x32_f16(
          false, A, false, Bf, (short)0, acc, false, false);
    }
    float bias = b2[nt * 16 + (lane & 15)];
#pragma unroll
    for (int r = 0; r < 8; ++r) {
      float h = fmaxf(acc[r] + bias, 0.f);                 // relu(h@W2+b2)
      out[(brow + mrb + r) * H2 + nt * 16 + (lane & 15)] = h;
    }
  }
}

// ---------------------------------------------------------------------------
// Launch: inputs in setup_inputs() order: x, W1, b1, W2, b2. Output (B, 32) f32.
// Workspace: W1p = 4*35*32*16 halfs (143,360 B), W2p = 2*2*32*16 halfs (4 KB).
// ---------------------------------------------------------------------------
extern "C" void kernel_launch(void* const* d_in, const int* in_sizes, int n_in,
                              void* d_out, int out_size, void* d_ws, size_t ws_size,
                              hipStream_t stream) {
  (void)in_sizes; (void)n_in; (void)out_size; (void)ws_size;
  const float* x  = (const float*)d_in[0];
  const float* W1 = (const float*)d_in[1];
  const float* b1 = (const float*)d_in[2];
  const float* W2 = (const float*)d_in[3];
  const float* b2 = (const float*)d_in[4];
  float* out = (float*)d_out;

  _Float16* W1p = (_Float16*)d_ws;
  _Float16* W2p = W1p + (size_t)(H1 / 16) * KB1 * 32 * 16;   // 71,680 halfs in

  const int packTasks = (H1 / 16) * KB1 * 32 + (H2 / 16) * 2 * 32; // 4608
  pack_weights<<<(packTasks + 255) / 256, 256, 0, stream>>>(W1, W2, W1p, W2p);

  pose_fused<<<BATCH / ROWS, 128, 0, stream>>>(x, b1, b2, W1p, W2p, out);
}
